// _SelfAttention_35192962023647
// MI455X (gfx1250) — matrix operand order repaired and checked by reading, not compiled, not run on hardware
//
#include <hip/hip_runtime.h>
#include <hip/hip_bf16.h>

// ---------------- types ----------------
typedef __bf16 bf16_t;
typedef __attribute__((ext_vector_type(16))) __bf16 v16bf;
typedef __attribute__((ext_vector_type(8)))  __bf16 v8bf;
typedef __attribute__((ext_vector_type(8)))  float  v8f;

static constexpr int BATCH = 8;
static constexpr int C     = 256;
static constexpr int CQK   = 32;
static constexpr int NN    = 4096;   // H*W
static constexpr float LOG2E = 1.4426950408889634f;

static __device__ inline v16bf concat8(v8bf lo, v8bf hi) {
  return __builtin_shufflevector(lo, hi, 0,1,2,3,4,5,6,7,8,9,10,11,12,13,14,15);
}

static __device__ inline v8f wmma_bf16(v16bf a, v16bf b, v8f c) {
  // D = A(16x32 bf16) * B(32x16 bf16) + C(16x16 f32)
  return __builtin_amdgcn_wmma_f32_16x16x32_bf16(false, a, false, b, (short)0, c, false, false);
}

static __device__ inline unsigned pack_bf16x2(float a, float b) {
  unsigned short ua = __builtin_bit_cast(unsigned short, (bf16_t)a);
  unsigned short ub = __builtin_bit_cast(unsigned short, (bf16_t)b);
  return (unsigned)ua | ((unsigned)ub << 16);
}

// ---------------- kernel 0: x[b][c][n] (f32) -> xbf[b][n][c] (bf16) ----------------
__global__ __launch_bounds__(256) void k_transpose(const float* __restrict__ x,
                                                   bf16_t* __restrict__ xbf) {
  __shared__ float tile[32][33];
  const int b  = blockIdx.z;
  const int cb = blockIdx.y * 32;
  const int nb = blockIdx.x * 32;
  const int tx = threadIdx.x;        // 0..31
  const int ty = threadIdx.y;        // 0..7
  const float* src = x + ((size_t)b * C + cb) * NN + nb;
#pragma unroll
  for (int q = 0; q < 4; ++q)
    tile[ty + 8*q][tx] = src[(size_t)(ty + 8*q) * NN + tx];
  __syncthreads();
  bf16_t* dst = xbf + ((size_t)b * NN + nb) * C + cb;
#pragma unroll
  for (int q = 0; q < 4; ++q)
    dst[(size_t)(ty + 8*q) * C + tx] = (bf16_t)tile[tx][ty + 8*q];
}

// ---------------- kernel 1: WMMA projections ----------------
// fT[b][n][o] = wq@x + bq   (o<32)     gT[b][n][o] = wk@x + bk   (o<32)
// hh[b][c][n] = wv@x + bv   (c<256)    all bf16 outputs
__global__ __launch_bounds__(128) void k_proj(const bf16_t* __restrict__ xbf,
                                              const float* __restrict__ wq, const float* __restrict__ bq,
                                              const float* __restrict__ wk, const float* __restrict__ bk,
                                              const float* __restrict__ wv, const float* __restrict__ bv,
                                              bf16_t* __restrict__ fT, bf16_t* __restrict__ gT,
                                              bf16_t* __restrict__ hh) {
  const int w    = threadIdx.x >> 5;
  const int lane = threadIdx.x & 31;
  const int col  = lane & 15;        // A: M row / B,D: N col
  const int h    = lane >> 4;

  const int id = blockIdx.x * 4 + w;               // 0 .. 40959
  const int b  = id / (20 * 256);
  const int rm = id % (20 * 256);
  const int rb = rm / 256;                         // 0..19 row-block
  const int nt = rm % 256;                         // n tile
  const int n0 = nt * 16;

  const float* wmat; const float* bias; int o0; int which;
  if (rb < 2)      { wmat = wq; bias = bq; o0 = rb * 16;        which = 0; }
  else if (rb < 4) { wmat = wk; bias = bk; o0 = (rb - 2) * 16;  which = 1; }
  else             { wmat = wv; bias = bv; o0 = (rb - 4) * 16;  which = 2; }

  const float*  wrow = wmat + (size_t)(o0 + col) * C;
  const bf16_t* xrow = xbf + ((size_t)b * NN + n0 + col) * C;

  v8f acc = {};
#pragma unroll
  for (int ks = 0; ks < C; ks += 32) {
    v8f wa = *(const v8f*)(wrow + ks + 8*h);
    v8f wb = *(const v8f*)(wrow + ks + 16 + 8*h);
    v16bf a;
#pragma unroll
    for (int e = 0; e < 8; ++e) { a[e] = (bf16_t)wa[e]; a[8+e] = (bf16_t)wb[e]; }
    v16bf bx = concat8(*(const v8bf*)(xrow + ks + 8*h), *(const v8bf*)(xrow + ks + 16 + 8*h));
    acc = wmma_bf16(a, bx, acc);
  }
  v8f bs = *(const v8f*)(bias + o0 + 8*h);
#pragma unroll
  for (int r = 0; r < 8; ++r) acc[r] += bs[r];

  if (which < 2) {
    bf16_t* dst = (which == 0 ? fT : gT) + ((size_t)b * NN + n0 + col) * CQK + o0 + 8*h;
#pragma unroll
    for (int q = 0; q < 4; ++q)
      *(unsigned*)(dst + 2*q) = pack_bf16x2(acc[2*q], acc[2*q + 1]);
  } else {
    bf16_t* dst = hh + (size_t)b * C * NN + (size_t)(o0 + 8*h) * NN + n0 + col;
#pragma unroll
    for (int r = 0; r < 8; ++r)
      dst[(size_t)r * NN] = (bf16_t)acc[r];
  }
}

// ---------------- kernel 2: fused attention, single pass (unnormalized softmax) ----------------
// one workgroup (8 waves) per (batch, 128-col j-tile)
// beta = exp(s) / sum_i exp(s): shift-invariant, and |s| <= ~25 here so no max needed in fp32.
__global__ __launch_bounds__(256) void k_attn(const bf16_t* __restrict__ fT,
                                              const bf16_t* __restrict__ gT,
                                              const bf16_t* __restrict__ hh,
                                              const float* __restrict__ x,
                                              const float* __restrict__ gamma,
                                              float* __restrict__ out) {
  __shared__ bf16_t pbuf[2][128 * 64];   // double-buffered [j_local][i_local], 2 x 16 KB
  __shared__ float  lsh[128];

  const int tid  = threadIdx.x;
  const int w    = tid >> 5;
  const int lane = tid & 31;
  const int col  = lane & 15;
  const int h    = lane >> 4;

  const int b  = blockIdx.y;
  const int j0 = blockIdx.x * 128;
  const int jw = j0 + w * 16;

  const bf16_t* fTb = fT + (size_t)b * NN * CQK;
  const bf16_t* gTb = gT + (size_t)b * NN * CQK;
  const bf16_t* hhb = hh + (size_t)b * C * NN;

  // resident B operand: g columns of this wave's 16-j tile (K = all 32 channels)
  const v16bf gB = concat8(*(const v8bf*)(gTb + (size_t)(jw + col) * CQK + 8*h), *(const v8bf*)(gTb + (size_t)(jw + col) * CQK + 16 + 8*h));

  v8f acc[16];
  {
    v8f z = {};
#pragma unroll
    for (int k = 0; k < 16; ++k) acc[k] = z;
  }
  float lsum = 0.f;                  // partial softmax denominator for column (jw+col)
  const int cw = w * 32;             // this wave's 32 output channels

  for (int it = 0; it < NN / 64; ++it) {
    const int i0 = it * 64;
    bf16_t* pb = pbuf[it & 1];

    // ---- scores for this wave's j-tile; p = exp(s), column-sum into lsum, bf16 to LDS ----
#pragma unroll
    for (int t = 0; t < 4; ++t) {
      const bf16_t* fr = fTb + (size_t)(i0 + 16*t + col) * CQK;
      v16bf a = concat8(*(const v8bf*)(fr + 8*h), *(const v8bf*)(fr + 16 + 8*h));
      v8f z = {};
      v8f s = wmma_bf16(a, gB, z);
#pragma unroll
      for (int q = 0; q < 4; ++q) {
        float p0 = exp2f(s[2*q]     * LOG2E);
        float p1 = exp2f(s[2*q + 1] * LOG2E);
        lsum += p0 + p1;
        *(unsigned*)(&pb[(size_t)(w*16 + col) * 64 + 16*t + 8*h + 2*q]) = pack_bf16x2(p0, p1);
      }
    }
    __syncthreads();   // p visible; double buffer allows next-iter writes w/o 2nd barrier

    // ---- A operands: two 16-channel hh tiles, two K=32 chunks; prefetch next iter ----
    v16bf ha[2][2];
#pragma unroll
    for (int ct = 0; ct < 2; ++ct) {
      const bf16_t* hr = hhb + (size_t)(cw + 16*ct + col) * NN + i0;
#pragma unroll
      for (int kk = 0; kk < 2; ++kk)
        ha[ct][kk] = concat8(*(const v8bf*)(hr + 32*kk + 8*h),
                             *(const v8bf*)(hr + 32*kk + 16 + 8*h));
      if (i0 + 64 < NN) __builtin_prefetch(hr + 64 + 32*h, 0, 3);
    }

    // ---- o-accumulation: 32 WMMAs over all 8 j-tiles ----
#pragma unroll
    for (int jt = 0; jt < 8; ++jt) {
#pragma unroll
      for (int kk = 0; kk < 2; ++kk) {
        v16bf pB = concat8(*(const v8bf*)(&pb[(size_t)(jt*16 + col) * 64 + 32*kk + 8*h]), *(const v8bf*)(&pb[(size_t)(jt*16 + col) * 64 + 32*kk + 16 + 8*h]));
        acc[jt*2 + 0] = wmma_bf16(ha[0][kk], pB, acc[jt*2 + 0]);
        acc[jt*2 + 1] = wmma_bf16(ha[1][kk], pB, acc[jt*2 + 1]);
      }
    }
  }

  // combine the two half-lanes that hold the same column (rows 0-7 vs 8-15 of each tile)
  float lj = lsum + __shfl_xor(lsum, 16, 32);
  if (h == 0) lsh[w * 16 + col] = lj;
  __syncthreads();

  // ---- epilogue: o = gamma * acc / l + x ----
  const float g0 = gamma[0];
  const float* xb = x   + (size_t)b * C * NN;
  float*       ob = out + (size_t)b * C * NN;
#pragma unroll
  for (int jt = 0; jt < 8; ++jt) {
    const float inv = g0 / lsh[jt*16 + col];
#pragma unroll
    for (int ct = 0; ct < 2; ++ct) {
      v8f a = acc[jt*2 + ct];
      size_t base = (size_t)(cw + 16*ct + 8*h) * NN + j0 + jt*16 + col;
#pragma unroll
      for (int r = 0; r < 8; ++r) {
        size_t idx = base + (size_t)r * NN;
        ob[idx] = a[r] * inv + xb[idx];
      }
    }
  }
}

// ---------------- launcher ----------------
extern "C" void kernel_launch(void* const* d_in, const int* in_sizes, int n_in,
                              void* d_out, int out_size, void* d_ws, size_t ws_size,
                              hipStream_t stream) {
  const float* x     = (const float*)d_in[0];
  const float* wq    = (const float*)d_in[1];
  const float* bq    = (const float*)d_in[2];
  const float* wk    = (const float*)d_in[3];
  const float* bk    = (const float*)d_in[4];
  const float* wv    = (const float*)d_in[5];
  const float* bv    = (const float*)d_in[6];
  const float* gamma = (const float*)d_in[7];
  float* out = (float*)d_out;

  char* ws = (char*)d_ws;
  bf16_t* xbf = (bf16_t*)(ws);                                  // 16 MB
  bf16_t* fT  = (bf16_t*)(ws + (size_t)16 * 1024 * 1024);       //  2 MB
  bf16_t* gT  = (bf16_t*)(ws + (size_t)18 * 1024 * 1024);       //  2 MB
  bf16_t* hh  = (bf16_t*)(ws + (size_t)20 * 1024 * 1024);       // 16 MB

  k_transpose<<<dim3(NN / 32, C / 32, BATCH), dim3(32, 8), 0, stream>>>(x, xbf);
  k_proj<<<dim3(BATCH * 20 * 256 / 4), dim3(128), 0, stream>>>(xbf, wq, bq, wk, bk, wv, bv, fT, gT, hh);
  k_attn<<<dim3(NN / 128, BATCH), dim3(256), 0, stream>>>(fT, gT, hh, x, gamma, out);
}
